// PixtralHFVisionModel_80401787781164
// MI455X (gfx1250) — compile-verified, matrix operand order checked by reading, not run on hardware
//
#include <hip/hip_runtime.h>
#include <hip/hip_bf16.h>

typedef _Float16 half_t;
typedef __attribute__((ext_vector_type(16))) _Float16 v16h;
typedef __attribute__((ext_vector_type(8)))  _Float16 v8h;
typedef __attribute__((ext_vector_type(8)))  float    v8f;

#define HIDN   1024
#define NHEADS 16
#define HDIM   64
#define INTERN 4096
#define NLAY   4
#define STOK   1536
#define S_IMG0 1024

// GEMM block tiling
#define BM 128
#define BN 128
#define BK 32
#define LDT 40  // padded LDS row pitch (halves): 40*2B = 80B -> conflict-free

// ---------------------------------------------------------------------------
// CDNA5 async global->LDS copy (ASYNCcnt-tracked), 16B per lane.
// Inline asm: VDST = VGPR with LDS byte address (generic shared pointers
// truncate to addr[31:0] == LDS offset per ISA aperture rules); VADDR = 64-bit
// global address.
// ---------------------------------------------------------------------------
__device__ __forceinline__ void async_copy16(const half_t* g, half_t* l) {
  unsigned int loff = (unsigned int)(uintptr_t)l;          // LDS byte offset
  unsigned long long ga = (unsigned long long)(uintptr_t)g;
  asm volatile("global_load_async_to_lds_b128 %0, %1, off"
               :: "v"(loff), "v"(ga) : "memory");
}

__device__ __forceinline__ void async_wait4() {
  asm volatile("s_wait_asynccnt 0x4" ::: "memory");
}
__device__ __forceinline__ void async_wait0() {
  asm volatile("s_wait_asynccnt 0x0" ::: "memory");
}

// ---------------------------------------------------------------------------
// WMMA helpers (CDNA5 wave32, 16x16x32 f16 -> f32)
// ---------------------------------------------------------------------------
__device__ __forceinline__ v8f wmma_f16(v16h a, v16h b, v8f c) {
  return __builtin_amdgcn_wmma_f32_16x16x32_f16(false, a, false, b, (short)0, c,
                                                false, false);
}

// 16x32 f16 fragment from row-major global memory.
// ISA layout: lanes 0-15 hold row r%16, K {0..7,16..23}; lanes 16-31 K {8..15,24..31}.
__device__ __forceinline__ v16h load_frag(const half_t* __restrict__ p, int ld,
                                          int row, int k, int lane) {
  const half_t* base =
      p + (size_t)(row + (lane & 15)) * (size_t)ld + (k + ((lane >> 4) << 3));
  v8h lo = *(const v8h*)(base);
  v8h hi = *(const v8h*)(base + 16);
  v16h r;
#pragma unroll
  for (int i = 0; i < 8; ++i) { r[i] = lo[i]; r[i + 8] = hi[i]; }
  return r;
}

// Same fragment layout, but from a staged LDS tile (pitch LDT, k-base 0).
__device__ __forceinline__ v16h lds_frag(const half_t* p, int row, int lane) {
  const half_t* base = p + (row + (lane & 15)) * LDT + ((lane >> 4) << 3);
  v8h lo = *(const v8h*)(base);
  v8h hi = *(const v8h*)(base + 16);
  v16h r;
#pragma unroll
  for (int i = 0; i < 8; ++i) { r[i] = lo[i]; r[i + 8] = hi[i]; }
  return r;
}

// ---------------------------------------------------------------------------
// LDS-staged, double-buffered WMMA GEMM:  C[M,N] = A[M,K] * W[N,K]^T
// Block 256 thr = 8 waves, tile 128x128; wave tile 32x64 (8 WMMAs / k-step).
// A/B K-tiles staged via async global->LDS (ASYNCcnt), double buffered.
// mode 0: f32 out (+opt residual); 1: f16 out; 2: f16 transposed (V^T build).
// All dims are multiples of tile sizes -> no bounds checks.
// ---------------------------------------------------------------------------
__global__ __launch_bounds__(256) void gemm_kernel(
    const half_t* __restrict__ A, int lda, const half_t* __restrict__ W,
    int ldw, float* __restrict__ Cf, half_t* __restrict__ Ch,
    const float* __restrict__ resid, int ldc, int K, int mode) {
  __shared__ half_t sA[2][BM * LDT];
  __shared__ half_t sB[2][BN * LDT];

  const int t = threadIdx.x;
  const int lane = t & 31;
  const int wid = t >> 5;
  const int gm0 = blockIdx.x * BM;
  const int gn0 = blockIdx.y * BN;
  const int m0 = (wid & 3) * 32;   // 4 waves along M
  const int n0 = (wid >> 2) * 64;  // 2 waves along N

  // staging map: 512 16B chunks per tile; thread handles rows r0 and r0+64
  const int r0 = t >> 2;
  const int c0 = (t & 3) * 8;  // halves

  auto stage = [&](int buf, int kk) {
    const half_t* ga0 = A + (size_t)(gm0 + r0) * lda + kk + c0;
    const half_t* ga1 = A + (size_t)(gm0 + r0 + 64) * lda + kk + c0;
    const half_t* gb0 = W + (size_t)(gn0 + r0) * ldw + kk + c0;
    const half_t* gb1 = W + (size_t)(gn0 + r0 + 64) * ldw + kk + c0;
    async_copy16(ga0, &sA[buf][r0 * LDT + c0]);
    async_copy16(ga1, &sA[buf][(r0 + 64) * LDT + c0]);
    async_copy16(gb0, &sB[buf][r0 * LDT + c0]);
    async_copy16(gb1, &sB[buf][(r0 + 64) * LDT + c0]);
  };

  v8f acc[2][4] = {};
  const int KT = K / BK;
  stage(0, 0);
  for (int kt = 0; kt < KT; ++kt) {
    if (kt + 1 < KT) {
      stage((kt + 1) & 1, (kt + 1) * BK);
      async_wait4();  // current stage done; only next stage (4 ops) in flight
    } else {
      async_wait0();
    }
    __syncthreads();
    const half_t* pa = sA[kt & 1];
    const half_t* pb = sB[kt & 1];
    v16h a0 = lds_frag(pa, m0, lane);
    v16h a1 = lds_frag(pa, m0 + 16, lane);
    v16h b0 = lds_frag(pb, n0, lane);
    v16h b1 = lds_frag(pb, n0 + 16, lane);
    v16h b2 = lds_frag(pb, n0 + 32, lane);
    v16h b3 = lds_frag(pb, n0 + 48, lane);
    acc[0][0] = wmma_f16(a0, b0, acc[0][0]);
    acc[0][1] = wmma_f16(a0, b1, acc[0][1]);
    acc[0][2] = wmma_f16(a0, b2, acc[0][2]);
    acc[0][3] = wmma_f16(a0, b3, acc[0][3]);
    acc[1][0] = wmma_f16(a1, b0, acc[1][0]);
    acc[1][1] = wmma_f16(a1, b1, acc[1][1]);
    acc[1][2] = wmma_f16(a1, b2, acc[1][2]);
    acc[1][3] = wmma_f16(a1, b3, acc[1][3]);
    __syncthreads();  // protect buffer being restaged next iteration
  }

  const int hb = (lane >> 4) << 3;  // +8 rows for upper half-lanes (C layout)
#pragma unroll
  for (int ti = 0; ti < 2; ++ti) {
#pragma unroll
    for (int tj = 0; tj < 4; ++tj) {
      const v8f a = acc[ti][tj];
      const int col = gn0 + n0 + tj * 16 + (lane & 15);
#pragma unroll
      for (int e = 0; e < 8; ++e) {
        const int row = gm0 + m0 + ti * 16 + hb + e;
        const float v = a[e];
        if (mode == 0) {
          const size_t idx = (size_t)row * ldc + col;
          Cf[idx] = (resid ? resid[idx] : 0.f) + v;
        } else if (mode == 1) {
          Ch[(size_t)row * ldc + col] = (half_t)v;
        } else {
          Ch[(size_t)col * ldc + row] = (half_t)v;  // transposed store
        }
      }
    }
  }
}

// ---------------------------------------------------------------------------
// RMSNorm over HID=1024 columns; one block (256 thr) per row.
// ---------------------------------------------------------------------------
__global__ __launch_bounds__(256) void rms_kernel(const float* __restrict__ x,
                                                  const float* __restrict__ w,
                                                  float* __restrict__ of,
                                                  half_t* __restrict__ oh) {
  const int row = blockIdx.x;
  const float* xr = x + (size_t)row * HIDN;
  float v[4];
  float ss = 0.f;
#pragma unroll
  for (int i = 0; i < 4; ++i) {
    v[i] = xr[threadIdx.x + 256 * i];
    ss += v[i] * v[i];
  }
#pragma unroll
  for (int off = 16; off; off >>= 1) ss += __shfl_xor(ss, off, 32);
  __shared__ float ps[8];
  if ((threadIdx.x & 31) == 0) ps[threadIdx.x >> 5] = ss;
  __syncthreads();
  float tot = 0.f;
#pragma unroll
  for (int i = 0; i < 8; ++i) tot += ps[i];
  const float r = rsqrtf(tot * (1.0f / HIDN) + 1e-5f);
#pragma unroll
  for (int i = 0; i < 4; ++i) {
    const int c = threadIdx.x + 256 * i;
    const float y = v[i] * r * w[c];
    if (of) of[(size_t)row * HIDN + c] = y;
    else    oh[(size_t)row * HIDN + c] = (half_t)y;
  }
}

// ---------------------------------------------------------------------------
// im2col: f16 A matrix [STOK, 768] matching conv_w [HID][3*16*16] layout
// ---------------------------------------------------------------------------
__global__ __launch_bounds__(256) void im2col_kernel(
    const float* __restrict__ img0, const float* __restrict__ img1,
    half_t* __restrict__ out) {
  const int idx = blockIdx.x * 256 + threadIdx.x;  // over 1536*768
  const int s = idx / 768, j = idx % 768;
  const int c = j >> 8, py = (j >> 4) & 15, px = j & 15;
  float v;
  if (s < S_IMG0) {
    const int h = s >> 5, w = s & 31;  // 32x32 grid
    v = img0[((size_t)c * 512 + h * 16 + py) * 512 + (w * 16 + px)];
  } else {
    const int t = s - S_IMG0;
    const int h = t >> 4, w = t & 15;  // 32x16 grid
    v = img1[((size_t)c * 512 + h * 16 + py) * 256 + (w * 16 + px)];
  }
  out[idx] = (half_t)v;
}

// ---------------------------------------------------------------------------
// f32 -> f16 bulk convert (weights)
// ---------------------------------------------------------------------------
__global__ __launch_bounds__(256) void cvt_kernel(const float* __restrict__ in,
                                                  half_t* __restrict__ out,
                                                  int n) {
  const int i = blockIdx.x * 256 + threadIdx.x;
  if (i < n) out[i] = (half_t)in[i];
}

// ---------------------------------------------------------------------------
// Pixtral 2D RoPE: thread per (s, head, j), j in [0,32); rotates pair (j, j+32)
// ---------------------------------------------------------------------------
__global__ __launch_bounds__(256) void rope_kernel(const float* __restrict__ Q,
                                                   const float* __restrict__ K,
                                                   half_t* __restrict__ Qh,
                                                   half_t* __restrict__ Kh) {
  const int idx = blockIdx.x * 256 + threadIdx.x;  // 1536*16*32
  const int j = idx & 31;
  const int head = (idx >> 5) & 15;
  const int s = idx >> 9;
  int hp, wp;
  if (s < S_IMG0) { hp = s >> 5; wp = s & 31; }
  else            { const int t = s - S_IMG0; hp = t >> 4; wp = t & 15; }
  float angle;
  if (j < 16) angle = (float)hp * __powf(10000.f, -(4.f * j) / 64.f);
  else        angle = (float)wp * __powf(10000.f, -(4.f * (j - 16) + 2.f) / 64.f);
  float sn, cs;
  __sincosf(angle, &sn, &cs);
  const size_t base = (size_t)s * HIDN + head * HDIM;
  const float q0 = Q[base + j], q1 = Q[base + j + 32];
  Qh[base + j]      = (half_t)(q0 * cs - q1 * sn);
  Qh[base + j + 32] = (half_t)(q1 * cs + q0 * sn);
  const float k0 = K[base + j], k1 = K[base + j + 32];
  Kh[base + j]      = (half_t)(k0 * cs - k1 * sn);
  Kh[base + j + 32] = (half_t)(k1 * cs + k0 * sn);
}

// ---------------------------------------------------------------------------
// Flash attention, one wave per (head, 16-q-row tile).
// Scores computed TRANSPOSED (A = K rows, B = Q rows): the C-layout of
// scores^T is exactly the A-fragment layout for the PV WMMA -> no transpose.
// Block-diagonal mask == restricted k-range (image boundaries 32-aligned).
// ---------------------------------------------------------------------------
__global__ __launch_bounds__(32) void attn_kernel(
    const half_t* __restrict__ Qh, const half_t* __restrict__ Kh,
    const half_t* __restrict__ Vt, half_t* __restrict__ Oh) {
  const int head = blockIdx.y;
  const int q0 = blockIdx.x * 16;
  const int lane = threadIdx.x;
  const int kstart = (q0 < S_IMG0) ? 0 : S_IMG0;
  const int kend = (q0 < S_IMG0) ? S_IMG0 : STOK;

  const half_t* Qp = Qh + head * HDIM;                 // [S,64] slice, ld 1024
  const half_t* Kp = Kh + head * HDIM;
  const half_t* Vp = Vt + (size_t)head * HDIM * STOK;  // [64,S], ld STOK

  const v16h bq0 = load_frag(Qp, HIDN, q0, 0, lane);
  const v16h bq1 = load_frag(Qp, HIDN, q0, 32, lane);

  v8f out0 = {}, out1 = {}, out2 = {}, out3 = {};
  float m = -1e30f, l = 0.f;
  const int hb = (lane >> 4) << 3;

  for (int k = kstart; k < kend; k += 32) {
    v16h ak0a = load_frag(Kp, HIDN, k, 0, lane);
    v16h ak0b = load_frag(Kp, HIDN, k, 32, lane);
    v16h ak1a = load_frag(Kp, HIDN, k + 16, 0, lane);
    v16h ak1b = load_frag(Kp, HIDN, k + 16, 32, lane);
    v8f s0 = {}, s1 = {};
    s0 = wmma_f16(ak0a, bq0, s0);
    s0 = wmma_f16(ak0b, bq1, s0);
    s1 = wmma_f16(ak1a, bq0, s1);
    s1 = wmma_f16(ak1b, bq1, s1);

    float mx = m;
#pragma unroll
    for (int e = 0; e < 8; ++e) {
      s0[e] *= 0.125f;
      s1[e] *= 0.125f;
      mx = fmaxf(mx, fmaxf(s0[e], s1[e]));
    }
    mx = fmaxf(mx, __shfl_xor(mx, 16, 32));
    const float corr = __expf(m - mx);
    m = mx;

    v16h p;
    float psum = 0.f;
#pragma unroll
    for (int e = 0; e < 8; ++e) {
      const float p0 = __expf(s0[e] - mx);
      const float p1 = __expf(s1[e] - mx);
      psum += p0 + p1;
      p[e] = (half_t)p0;
      p[e + 8] = (half_t)p1;
    }
    l = l * corr + psum + __shfl_xor(psum, 16, 32);

    float cb[8];
#pragma unroll
    for (int e = 0; e < 8; ++e) cb[e] = __shfl(corr, e + hb, 32);
#pragma unroll
    for (int e = 0; e < 8; ++e) {
      out0[e] *= cb[e]; out1[e] *= cb[e]; out2[e] *= cb[e]; out3[e] *= cb[e];
    }

    out0 = wmma_f16(p, load_frag(Vp, STOK, 0, k, lane), out0);
    out1 = wmma_f16(p, load_frag(Vp, STOK, 16, k, lane), out1);
    out2 = wmma_f16(p, load_frag(Vp, STOK, 32, k, lane), out2);
    out3 = wmma_f16(p, load_frag(Vp, STOK, 48, k, lane), out3);
  }

  float lb[8];
#pragma unroll
  for (int e = 0; e < 8; ++e) lb[e] = 1.f / __shfl(l, e + hb, 32);
#pragma unroll
  for (int t = 0; t < 4; ++t) {
    const v8f o = (t == 0) ? out0 : (t == 1) ? out1 : (t == 2) ? out2 : out3;
    const int d = t * 16 + (lane & 15);
#pragma unroll
    for (int e = 0; e < 8; ++e) {
      const int q = q0 + hb + e;
      Oh[(size_t)q * HIDN + head * HDIM + d] = (half_t)(o[e] * lb[e]);
    }
  }
}

// ---------------------------------------------------------------------------
// ff = silu(gate) * up -> f16
// ---------------------------------------------------------------------------
__global__ __launch_bounds__(256) void silu_mul_kernel(
    const float* __restrict__ g, const float* __restrict__ u,
    half_t* __restrict__ o) {
  const int i = blockIdx.x * 256 + threadIdx.x;  // STOK*INTER, exact grid
  const float gv = g[i];
  const float s = gv / (1.f + __expf(-gv));
  o[i] = (half_t)(s * u[i]);
}

// ---------------------------------------------------------------------------
// Host orchestration
// ---------------------------------------------------------------------------
static inline void launch_gemm(hipStream_t st, const half_t* A, int lda,
                               const half_t* W, int ldw, float* Cf, half_t* Ch,
                               const float* resid, int M, int N, int ldc,
                               int K, int mode) {
  dim3 g(M / BM, N / BN), b(256);
  gemm_kernel<<<g, b, 0, st>>>(A, lda, W, ldw, Cf, Ch, resid, ldc, K, mode);
}

extern "C" void kernel_launch(void* const* d_in, const int* in_sizes, int n_in,
                              void* d_out, int out_size, void* d_ws,
                              size_t ws_size, hipStream_t stream) {
  (void)in_sizes; (void)n_in; (void)out_size; (void)ws_size;
  const float* img0 = (const float*)d_in[0];
  const float* img1 = (const float*)d_in[1];
  const float* conv_w = (const float*)d_in[2];
  const float* ln_pre_w = (const float*)d_in[3];
  const float* attn_norm_w = (const float*)d_in[4];
  const float* q_w = (const float*)d_in[5];
  const float* k_w = (const float*)d_in[6];
  const float* v_w = (const float*)d_in[7];
  const float* o_w = (const float*)d_in[8];
  const float* ffn_norm_w = (const float*)d_in[9];
  const float* gate_w = (const float*)d_in[10];
  const float* up_w = (const float*)d_in[11];
  const float* down_w = (const float*)d_in[12];

  char* wp = (char*)d_ws;
  auto alloc = [&](size_t bytes) -> void* {
    void* r = (void*)wp;
    wp += (bytes + 255) & ~(size_t)255;
    return r;
  };
  const size_t SH = (size_t)STOK * HIDN;
  const size_t SI = (size_t)STOK * INTERN;
  const size_t WQK = (size_t)NLAY * HIDN * HIDN;
  const size_t WFF = (size_t)NLAY * INTERN * HIDN;

  half_t* wc = (half_t*)alloc((size_t)HIDN * 768 * 2);
  half_t* wq = (half_t*)alloc(WQK * 2);
  half_t* wk = (half_t*)alloc(WQK * 2);
  half_t* wv = (half_t*)alloc(WQK * 2);
  half_t* wo = (half_t*)alloc(WQK * 2);
  half_t* wg = (half_t*)alloc(WFF * 2);
  half_t* wu = (half_t*)alloc(WFF * 2);
  half_t* wd = (half_t*)alloc(WFF * 2);
  half_t* Aemb = (half_t*)alloc((size_t)STOK * 768 * 2);
  float*  Xtmp = (float*)alloc(SH * 4);
  float*  H2   = (float*)alloc(SH * 4);
  half_t* hn   = (half_t*)alloc(SH * 2);
  float*  Qf   = (float*)alloc(SH * 4);
  float*  Kf   = (float*)alloc(SH * 4);
  half_t* Qh   = (half_t*)alloc(SH * 2);
  half_t* Kh   = (half_t*)alloc(SH * 2);
  half_t* Vt   = (half_t*)alloc(SH * 2);  // [HID rows][STOK cols]
  half_t* Ohb  = (half_t*)alloc(SH * 2);
  float*  Gf   = (float*)alloc(SI * 4);
  float*  Uf   = (float*)alloc(SI * 4);
  half_t* Fh   = (half_t*)alloc(SI * 2);

  float* X = (float*)d_out;  // residual stream lives directly in d_out

  auto cvt = [&](const float* in, half_t* out, size_t n) {
    cvt_kernel<<<dim3((unsigned)((n + 255) / 256)), dim3(256), 0, stream>>>(
        in, out, (int)n);
  };
  cvt(conv_w, wc, (size_t)HIDN * 768);
  cvt(q_w, wq, WQK); cvt(k_w, wk, WQK); cvt(v_w, wv, WQK); cvt(o_w, wo, WQK);
  cvt(gate_w, wg, WFF); cvt(up_w, wu, WFF); cvt(down_w, wd, WFF);

  im2col_kernel<<<dim3((STOK * 768) / 256), dim3(256), 0, stream>>>(img0, img1,
                                                                    Aemb);
  launch_gemm(stream, Aemb, 768, wc, 768, Xtmp, nullptr, nullptr, STOK, HIDN,
              HIDN, 768, 0);

  rms_kernel<<<dim3(STOK), dim3(256), 0, stream>>>(Xtmp, ln_pre_w, X, nullptr);

  for (int l = 0; l < NLAY; ++l) {
    const half_t* wql = wq + (size_t)l * HIDN * HIDN;
    const half_t* wkl = wk + (size_t)l * HIDN * HIDN;
    const half_t* wvl = wv + (size_t)l * HIDN * HIDN;
    const half_t* wol = wo + (size_t)l * HIDN * HIDN;
    const half_t* wgl = wg + (size_t)l * (size_t)INTERN * HIDN;
    const half_t* wul = wu + (size_t)l * (size_t)INTERN * HIDN;
    const half_t* wdl = wd + (size_t)l * (size_t)HIDN * INTERN;

    rms_kernel<<<dim3(STOK), dim3(256), 0, stream>>>(
        X, attn_norm_w + (size_t)l * HIDN, nullptr, hn);

    launch_gemm(stream, hn, HIDN, wql, HIDN, Qf, nullptr, nullptr, STOK, HIDN,
                HIDN, HIDN, 0);
    launch_gemm(stream, hn, HIDN, wkl, HIDN, Kf, nullptr, nullptr, STOK, HIDN,
                HIDN, HIDN, 0);
    launch_gemm(stream, hn, HIDN, wvl, HIDN, nullptr, Vt, nullptr, STOK, HIDN,
                STOK, HIDN, 2);

    rope_kernel<<<dim3((STOK * NHEADS * 32) / 256), dim3(256), 0, stream>>>(
        Qf, Kf, Qh, Kh);

    attn_kernel<<<dim3(STOK / 16, NHEADS), dim3(32), 0, stream>>>(Qh, Kh, Vt,
                                                                  Ohb);

    launch_gemm(stream, Ohb, HIDN, wol, HIDN, H2, nullptr, X, STOK, HIDN, HIDN,
                HIDN, 0);

    rms_kernel<<<dim3(STOK), dim3(256), 0, stream>>>(
        H2, ffn_norm_w + (size_t)l * HIDN, nullptr, hn);

    launch_gemm(stream, hn, HIDN, wgl, HIDN, Gf, nullptr, nullptr, STOK,
                INTERN, INTERN, HIDN, 0);
    launch_gemm(stream, hn, HIDN, wul, HIDN, Uf, nullptr, nullptr, STOK,
                INTERN, INTERN, HIDN, 0);
    silu_mul_kernel<<<dim3((unsigned)(SI / 256)), dim3(256), 0, stream>>>(Gf,
                                                                          Uf,
                                                                          Fh);
    launch_gemm(stream, Fh, INTERN, wdl, INTERN, X, nullptr, H2, STOK, HIDN,
                HIDN, INTERN, 0);
  }
}
